// GraphMultiAttention_901943132285
// MI455X (gfx1250) — matrix operand order repaired and checked by reading, not compiled, not run on hardware
//
#include <hip/hip_runtime.h>
#include <hip/hip_bf16.h>

// ---------------------------------------------------------------------------
// GAT (2-layer, 8-head) for MI455X / gfx1250.
// Compute-bound (~50 GFLOP vs ~70MB HBM traffic after bitmask packing), so all
// matmuls use v_wmma_f32_16x16x32_f16 (f16 in, f32 accumulate) — the CDNA5
// matrix path. Attention is streamed flash-style: exp-weights are generated
// directly in the WMMA A-operand register layout; softmax denominator is
// accumulated in-register and reduced with a wave32 shfl_xor(16).
// sched_barrier(0) pins the per-chunk load group ahead of the consume chain,
// so all B tiles issue as one clause (single latency wait) and the VALU
// weight generation overlaps the loads.
// ---------------------------------------------------------------------------

typedef __attribute__((ext_vector_type(16))) _Float16 v16h;
typedef __attribute__((ext_vector_type(8)))  _Float16 v8h;
typedef __attribute__((ext_vector_type(8)))  float    v8f;

#define N_NODES   4096
#define MASKW_ROW 128           // 4096 / 32
#define LOG2E     1.44269504088896340736f

// ---------------------------------------------------------------------------
// adj[int32 N x N] -> packed bitmask, one 32-bit word per 32 columns.
// wave32 ballot: 32 lanes read 32 consecutive ints, lane0 writes the word.
__global__ void pack_mask_kernel(const int* __restrict__ adj,
                                 unsigned* __restrict__ mw) {
  int tid  = blockIdx.x * blockDim.x + threadIdx.x;
  int lane = tid & 31;
  int w    = tid >> 5;
  unsigned long long b = __ballot(adj[((size_t)w << 5) + lane] > 0);
  if (lane == 0) mw[w] = (unsigned)b;
}

// ---------------------------------------------------------------------------
__global__ void cast_f16_kernel(const float* __restrict__ in,
                                _Float16* __restrict__ out, int n) {
  int i = blockIdx.x * blockDim.x + threadIdx.x;
  if (i < n) out[i] = (_Float16)in[i];
}

// ---------------------------------------------------------------------------
// Rearrange a row-major [B][K][Nm] matrix into WMMA B-operand tiles:
// out[b][K/32][Nm/16][lane][16halves], lane<16: B[k=tk*32+(e&7)+16*(e>>3)][n=tn*16+lane],
// lane>=16: k += 8. One contiguous v16h per lane per WMMA afterwards.
template<typename TIN>
__global__ void swizzleB_kernel(const TIN* __restrict__ in,
                                _Float16* __restrict__ out,
                                int K, int Nm, int total) {
  int tid = blockIdx.x * blockDim.x + threadIdx.x;
  if (tid >= total) return;
  int e    = tid & 15;
  int lane = (tid >> 4) & 31;
  int t2   = tid >> 9;
  int NFt  = Nm >> 4, KT = K >> 5;
  int tn   = t2 % NFt;  int t3 = t2 / NFt;
  int tk   = t3 % KT;   int b  = t3 / KT;
  int k = (tk << 5) + ((e >> 3) << 4) + ((lane >> 4) << 3) + (e & 7);
  int n = (tn << 4) + (lane & 15);
  out[tid] = (_Float16)in[(size_t)b * K * Nm + (size_t)k * Nm + n];
}

// ---------------------------------------------------------------------------
// Dense GEMM: O[f16, MxNm] = A[f16, MxK row-major] * Bs (pre-swizzled tiles).
// 4 waves / block, each wave owns 16 rows x (NT*16) cols, fp32 accumulate.
// sched_barrier keeps the whole load group ahead of the WMMA chain: distinct
// registers per tile, one clause of b128 loads, single staggered wait.
template<int NT>
__global__ __launch_bounds__(128)
void gemm16_kernel(const _Float16* __restrict__ A, int lda, int K, long ABatch,
                   const _Float16* __restrict__ Bs, int NFt, long BsBatch,
                   _Float16* __restrict__ O, int ldo, long OBatch) {
  A  += (size_t)blockIdx.z * ABatch;
  Bs += (size_t)blockIdx.z * BsBatch;
  O  += (size_t)blockIdx.z * OBatch;
  int lane = threadIdx.x & 31;
  int wave = threadIdx.x >> 5;
  int m0   = (blockIdx.x * 4 + wave) * 16;
  int n0t  = blockIdx.y * NT;                     // first 16-col tile index
  const _Float16* Arow =
      A + (size_t)(m0 + (lane & 15)) * lda + ((lane >> 4) << 3);
  v8f zero = {0.f,0.f,0.f,0.f,0.f,0.f,0.f,0.f};
  v8f c[NT];
  #pragma unroll
  for (int t = 0; t < NT; ++t) c[t] = zero;
  for (int kc = 0; kc < K; kc += 32) {
    // issue every load for this chunk before any WMMA consumes them
    union { v16h v; v8h h[2]; } au;            // A-operand: lane<16 rows 0..15,
    au.h[0] = *(const v8h*)(Arow + kc);        // halves = K {0..7,16..23}
    au.h[1] = *(const v8h*)(Arow + kc + 16);   // (lanes>=16: +8)
    const _Float16* bp =
        Bs + ((size_t)((kc >> 5) * NFt + n0t) << 9) + (lane << 4);
    v16h b[NT];
    #pragma unroll
    for (int t = 0; t < NT; ++t) b[t] = *(const v16h*)(bp + ((size_t)t << 9));
    __builtin_amdgcn_sched_barrier(0);         // loads above, WMMAs below
    #pragma unroll
    for (int t = 0; t < NT; ++t)
      c[t] = __builtin_amdgcn_wmma_f32_16x16x32_f16(
          false, au.v, false, b[t], (short)0, c[t], false, false);
  }
  #pragma unroll
  for (int t = 0; t < NT; ++t)
    #pragma unroll
    for (int r = 0; r < 8; ++r)
      O[(size_t)(m0 + r + ((lane >> 4) << 3)) * ldo + (n0t + t) * 16 +
        (lane & 15)] = (_Float16)c[t][r];
}

// ---------------------------------------------------------------------------
// src[b][m] = Wh[b][m][:].a[b][:F], dst = Wh.a[b][F:]; fp32 accumulate.
__global__ void srcdst_kernel(const _Float16* __restrict__ Wh,
                              const float* __restrict__ a,
                              float* __restrict__ src, float* __restrict__ dst,
                              int M, int F, int total) {
  int tid = blockIdx.x * blockDim.x + threadIdx.x;
  if (tid >= total) return;
  int b = tid / M;
  const _Float16* row = Wh + (size_t)tid * F;
  const float* ab = a + (size_t)b * 2 * F;
  float s = 0.f, d = 0.f;
  for (int f = 0; f < F; ++f) {
    float v = (float)row[f];
    s += v * ab[f];
    d += v * ab[F + f];
  }
  src[tid] = s;
  dst[tid] = d;
}

// ---------------------------------------------------------------------------
// Fused masked-softmax + (attn @ Wh) via WMMA.
// Block = 256 threads = 8 waves; each wave owns a 16-row tile; blockIdx.y
// selects a group of FT 16-col feature tiles. dst[] staged in 16KB LDS,
// pre-scaled by log2(e) so each weight is a single v_exp_f32 (exp2).
// Weight tile A = exp(leakyrelu(src_i+dst_j)) * maskbit, generated straight
// into the A-operand lane layout; denominator accumulated per lane (lanes l
// and l+16 carry row l, disjoint K halves) then reduced with shfl_xor(16).
// sched_barrier after the B loads: the exp/mask VALU chain runs while the
// tiles are in flight (XDL/VALU co-execution).
template<int FT, bool FINAL>
__global__ __launch_bounds__(256)
void gat_attn_kernel(const _Float16* __restrict__ Bs, long BsHead,
                     const float* __restrict__ src,
                     const float* __restrict__ dst, int srcHead,
                     const unsigned* __restrict__ mw,
                     int NFt, void* __restrict__ OutP, int ldo,
                     int colPerHead) {
  __shared__ __align__(16) float dstS[N_NODES];
  int head = blockIdx.z;
  Bs  += (size_t)head * BsHead;
  src += (size_t)head * srcHead;
  dst += (size_t)head * srcHead;
  int tid = threadIdx.x;
  #pragma unroll
  for (int e = 0; e < 16; ++e)
    dstS[e * 256 + tid] = dst[e * 256 + tid] * LOG2E;   // log2-domain
  __syncthreads();

  int lane = tid & 31, wave = tid >> 5;
  int hi8  = (lane >> 4) << 3;
  int i0   = blockIdx.x * 128 + wave * 16;
  int row  = i0 + (lane & 15);
  float s_i = src[row] * LOG2E;                         // log2-domain
  int fg0  = blockIdx.y * FT;
  const unsigned* mrow = mw + (size_t)row * MASKW_ROW;

  v8f zero = {0.f,0.f,0.f,0.f,0.f,0.f,0.f,0.f};
  v8f c[FT];
  #pragma unroll
  for (int t = 0; t < FT; ++t) c[t] = zero;
  float den = 0.f;

  for (int jt = 0; jt < MASKW_ROW; ++jt) {
    // 1) issue all B-tile loads first: latency hides behind weight generation
    const _Float16* bp =
        Bs + ((size_t)(jt * NFt + fg0) << 9) + (lane << 4);
    v16h b[FT];
    #pragma unroll
    for (int t = 0; t < FT; ++t) b[t] = *(const v16h*)(bp + ((size_t)t << 9));
    if (jt + 1 < MASKW_ROW)
      __builtin_prefetch(bp + ((size_t)NFt << 9), 0, 0); // next chunk's tiles
    __builtin_amdgcn_sched_barrier(0);          // loads stay above this point

    // 2) weight tile directly in A-operand layout; dst via 4x ds_load_b128
    unsigned m = mrow[jt];
    union { float4 q[4]; float f[16]; } dv;
    const float4* dp = (const float4*)&dstS[(jt << 5) + hi8];
    dv.q[0] = dp[0]; dv.q[1] = dp[1];   // jloc = hi8 + 0..7
    dv.q[2] = dp[4]; dv.q[3] = dp[5];   // jloc = hi8 + 16..23
    union { v16h v; _Float16 h[16]; } au;
    #pragma unroll
    for (int e = 0; e < 16; ++e) {
      int jloc = (e & 7) + ((e >> 3) << 4) + hi8;   // A-operand K mapping
      float tv = s_i + dv.f[e];
      tv = tv > 0.f ? tv : 0.2f * tv;               // leakyrelu(0.2)
      float w = ((m >> jloc) & 1u) ? __builtin_amdgcn_exp2f(tv) : 0.f;
      den += w;
      au.h[e] = (_Float16)w;
    }
    // 3) accumulate
    #pragma unroll
    for (int t = 0; t < FT; ++t)
      c[t] = __builtin_amdgcn_wmma_f32_16x16x32_f16(
          false, au.v, false, b[t], (short)0, c[t], false, false);
  }

  den += __shfl_xor(den, 16, 32);                 // combine K halves (wave32)
  float inv = den > 0.f ? 1.f / den : 0.f;

  #pragma unroll
  for (int r = 0; r < 8; ++r) {
    float invr = __shfl(inv, r + hi8, 32);        // denom of C-row r+hi8
    size_t orow = (size_t)(i0 + r + hi8);
    #pragma unroll
    for (int t = 0; t < FT; ++t) {
      float v = c[t][r] * invr;
      v = v > 0.f ? v : (__expf(v) - 1.f);        // elu
      size_t ocol =
          (size_t)head * colPerHead + (size_t)(fg0 + t) * 16 + (lane & 15);
      if (FINAL)
        ((float*)OutP)[orow * (size_t)ldo + ocol] = v;
      else
        ((_Float16*)OutP)[orow * (size_t)ldo + ocol] = (_Float16)v;
    }
  }
}

// ---------------------------------------------------------------------------
extern "C" void kernel_launch(void* const* d_in, const int* in_sizes, int n_in,
                              void* d_out, int out_size, void* d_ws,
                              size_t ws_size, hipStream_t stream) {
  (void)in_sizes; (void)n_in; (void)out_size; (void)ws_size;
  const float* x      = (const float*)d_in[0];
  const int*   adj    = (const int*)d_in[1];
  // d_in[2] (adaptive_weight) is unused by the reference.
  const float* W_h    = (const float*)d_in[3];
  const float* a_h    = (const float*)d_in[4];
  const float* W_out  = (const float*)d_in[5];
  const float* a_out  = (const float*)d_in[6];

  char* ws = (char*)d_ws;
  size_t off = 0;
  auto alloc = [&](size_t bytes) -> void* {
    off = (off + 255) & ~(size_t)255;
    void* p = ws + off;
    off += bytes;
    return p;
  };
  _Float16* xh    = (_Float16*)alloc((size_t)4096 * 768 * 2);
  unsigned* maskw = (unsigned*)alloc((size_t)4096 * 128 * 4);
  _Float16* WhsW  = (_Float16*)alloc((size_t)8 * 768 * 64 * 2);  // W_h swizzled
  _Float16* WoutS = (_Float16*)alloc((size_t)512 * 768 * 2);     // W_out swizzled
  _Float16* Wh    = (_Float16*)alloc((size_t)8 * 4096 * 64 * 2); // [H][N][64]
  _Float16* WhT   = (_Float16*)alloc((size_t)8 * 4096 * 64 * 2); // swizzled
  float*    src1  = (float*)alloc((size_t)8 * 4096 * 4);
  float*    dst1  = (float*)alloc((size_t)8 * 4096 * 4);
  _Float16* hmat  = (_Float16*)alloc((size_t)4096 * 512 * 2);    // layer-1 out
  _Float16* Wh2   = (_Float16*)alloc((size_t)4096 * 768 * 2);
  _Float16* Wh2T  = (_Float16*)alloc((size_t)4096 * 768 * 2);
  float*    src2  = (float*)alloc((size_t)4096 * 4);
  float*    dst2  = (float*)alloc((size_t)4096 * 4);

  // 0) pack adjacency into bitmask (64MB -> 2MB, wave32 ballot)
  pack_mask_kernel<<<dim3((4096 * 4096 / 32) / 256), 256, 0, stream>>>(adj, maskw);
  // 1) x -> f16
  cast_f16_kernel<<<dim3((4096 * 768 + 255) / 256), 256, 0, stream>>>(
      x, xh, 4096 * 768);
  // 2) swizzle weights into B-operand tiles
  swizzleB_kernel<float><<<dim3((8 * 768 * 64 + 255) / 256), 256, 0, stream>>>(
      W_h, WhsW, 768, 64, 8 * 768 * 64);
  swizzleB_kernel<float><<<dim3((512 * 768 + 255) / 256), 256, 0, stream>>>(
      W_out, WoutS, 512, 768, 512 * 768);
  // 3) Wh[h] = x @ W_h[h]  (per-head 4096x768x64, WMMA)
  gemm16_kernel<4><<<dim3(64, 1, 8), 128, 0, stream>>>(
      xh, 768, 768, 0L, WhsW, 4, (long)(24 * 4 * 512), Wh, 64,
      (long)4096 * 64);
  // 4) swizzle Wh into B tiles for the attention GEMM
  swizzleB_kernel<_Float16><<<dim3((8 * 4096 * 64 + 255) / 256), 256, 0,
                              stream>>>(Wh, WhT, 4096, 64, 8 * 4096 * 64);
  // 5) src/dst attention logits
  srcdst_kernel<<<dim3((8 * 4096 + 255) / 256), 256, 0, stream>>>(
      Wh, a_h, src1, dst1, 4096, 64, 8 * 4096);
  // 6) layer-1 fused masked-softmax + attn@Wh, elu, concat heads -> hmat f16
  gat_attn_kernel<4, false><<<dim3(32, 1, 8), 256, 0, stream>>>(
      WhT, (long)(128 * 4 * 512), src1, dst1, 4096, maskw, 4, hmat, 512, 64);
  // 7) Wh2 = hmat @ W_out (4096x512x768, WMMA)
  gemm16_kernel<8><<<dim3(64, 6, 1), 128, 0, stream>>>(
      hmat, 512, 512, 0L, WoutS, 48, 0L, Wh2, 768, 0L);
  // 8) swizzle Wh2, src2/dst2
  swizzleB_kernel<_Float16><<<dim3((4096 * 768 + 255) / 256), 256, 0,
                              stream>>>(Wh2, Wh2T, 4096, 768, 4096 * 768);
  srcdst_kernel<<<dim3((4096 + 255) / 256), 256, 0, stream>>>(
      Wh2, a_out, src2, dst2, 4096, 768, 4096);
  // 9) layer-2 fused masked-softmax + attn@Wh2, elu -> d_out fp32
  gat_attn_kernel<8, true><<<dim3(32, 6, 1), 256, 0, stream>>>(
      Wh2T, 0L, src2, dst2, 0, maskw, 48, d_out, 768, 0);
}
